// CrossEfficientAttention_35184372089344
// MI455X (gfx1250) — matrix operand order repaired and checked by reading, not compiled, not run on hardware
//
#include <hip/hip_runtime.h>
#include <hip/hip_bf16.h>

// ---------------------------------------------------------------------------
// CrossEfficientAttention on MI455X (gfx1250, wave32, WMMA bf16)
//
// B=8, C=256, H=W=64 -> N=4096, heads=4, d=64.
// Pipeline: QGEMM+softmax_d, KVGEMM, softmax_N(k), context=k*V^T,
// Mmat=wo*context (folds wo out of the big pass), out=Mmat*q+bo.
// Memory-bound (~220MB @ 23.3TB/s); bf16 WMMA for all matmuls, f32 accum.
// All LDS staging now uses packed v8bf -> ds_store_b128 (no scalar b16 stores).
// ---------------------------------------------------------------------------

typedef __bf16 bf16_t;
typedef __attribute__((ext_vector_type(16))) __bf16 v16bf;
typedef __attribute__((ext_vector_type(8)))  __bf16 v8bf;
typedef __attribute__((ext_vector_type(8)))  float  v8f;
typedef long long i64;

union AFrag { v16bf v; v8bf h[2]; };

static constexpr int Bb   = 8;
static constexpr int Cc   = 256;
static constexpr int Nn   = 4096;   // H*W
static constexpr int KP   = 40;     // LDS row pitch in bf16 (32 + 8 pad, 80B rows)

__device__ __forceinline__ v8bf pack8(const float* v) {
    v8bf r;
#pragma unroll
    for (int i = 0; i < 8; ++i) r[i] = (bf16_t)v[i];
    return r;
}

// ---------------------------------------------------------------------------
// Generic batched GEMM: Y[b, m0+.., n0+..] = sum_k A[b,m,k] * B(...) (+bias)
//   A: (M x K) row-major, TA in {f32, bf16}, per-batch stride sAb.
//   B: if TRANSB: (Ncols x K) row-major  (context GEMM: B = V, "NT" form)
//      else     : (K x Ncols) row-major  (activation GEMMs)
//   64x64 output tile per 128-thread block; 4 waves x (16 rows x 64 cols);
//   K-step 32 via v_wmma_f32_16x16x32_bf16 on the documented lane layouts.
//   SOFTMAX_D: epilogue = scale + softmax over the 64-row tile (one head).
// ---------------------------------------------------------------------------
template <typename TA, typename TB, bool TRANSB, bool SOFTMAX_D,
          bool WRITE_F32, bool WRITE_BF16>
__global__ __launch_bounds__(128)
void gemm_wmma(const TA* __restrict__ Ag, i64 sAb, int lda,
               const TB* __restrict__ Bg, i64 sBb, int ldb,
               const float* __restrict__ bias,
               float* __restrict__ Yf, bf16_t* __restrict__ Ybf,
               i64 sYb, int ldy, int K, float scale, int bfMinRow)
{
    __shared__ __align__(16) bf16_t lds_a[64 * KP];
    __shared__ __align__(16) bf16_t lds_b[64 * KP];

    const int tid   = threadIdx.x;
    const int wave  = tid >> 5;
    const int lane  = tid & 31;
    const int lrow  = lane & 15;
    const int lhalf = lane >> 4;

    const int m0 = blockIdx.y * 64;
    const int n0 = blockIdx.x * 64;
    const int b  = blockIdx.z;

    const TA* Ab  = Ag + (i64)b * sAb;
    const TB* Bg2 = Bg + (i64)b * sBb;

    v8f acc[4] = {};

    for (int k0 = 0; k0 < K; k0 += 32) {
        // ---- stage A tile: 64 rows x 32 K -> bf16, 2x ds_store_b128/thread --
        {
            const int r     = tid >> 1;
            const int cbase = (tid & 1) * 16;
            const TA* src = Ab + (i64)(m0 + r) * lda + k0 + cbase;
            __builtin_prefetch((const void*)(src + 32), 0, 1); // next K-step
            float tmp[16];
#pragma unroll
            for (int i = 0; i < 16; ++i) tmp[i] = (float)src[i];
            v8bf* dst = (v8bf*)&lds_a[r * KP + cbase];
            dst[0] = pack8(tmp);
            dst[1] = pack8(tmp + 8);
        }
        // ---- stage B tile into (n, k) layout ----
        if constexpr (TRANSB) {
            // B rows are contiguous in K: straight copy, vectorized stores.
            const int r     = tid >> 1;
            const int cbase = (tid & 1) * 16;
            const TB* src = Bg2 + (i64)(n0 + r) * ldb + k0 + cbase;
            __builtin_prefetch((const void*)(src + 32), 0, 1);
            float tmp[16];
#pragma unroll
            for (int i = 0; i < 16; ++i) tmp[i] = (float)src[i];
            v8bf* dst = (v8bf*)&lds_b[r * KP + cbase];
            dst[0] = pack8(tmp);
            dst[1] = pack8(tmp + 8);
        } else {
            // Transpose-on-stage: each thread owns one LDS row (fixed n,
            // 16 consecutive k). Global reads are stride-ldb b32 loads but
            // fully coalesced across the 64 consecutive-n threads; LDS side
            // is two aligned ds_store_b128 instead of 16 scalar b16 stores.
            const int nl    = tid & 63;
            const int khalf = (tid >> 6) * 16;   // 0 or 16
            const TB* src = Bg2 + (i64)(k0 + khalf) * ldb + n0 + nl;
            __builtin_prefetch((const void*)(src + (i64)32 * ldb), 0, 1);
            float tmp[16];
#pragma unroll
            for (int i = 0; i < 16; ++i) tmp[i] = (float)src[(i64)i * ldb];
            v8bf* dst = (v8bf*)&lds_b[nl * KP + khalf];
            dst[0] = pack8(tmp);
            dst[1] = pack8(tmp + 8);
        }
        __syncthreads();

        // ---- A fragment (16x32, documented 16-bit A layout) ----
        AFrag af;
        {
            const bf16_t* base = &lds_a[(wave * 16 + lrow) * KP + lhalf * 8];
            af.h[0] = *(const v8bf*)(base);       // K k0+{0..7}   / {8..15}
            af.h[1] = *(const v8bf*)(base + 16);  // K k0+{16..23} / {24..31}
        }
        // ---- 4 B fragments + WMMA ----
#pragma unroll
        for (int t = 0; t < 4; ++t) {
            AFrag bfr;
            const bf16_t* base = &lds_b[(t * 16 + lrow) * KP + lhalf * 8];
            bfr.h[0] = *(const v8bf*)(base);       // K k0+{0..7}   / {8..15}
            bfr.h[1] = *(const v8bf*)(base + 16);  // K k0+{16..23} / {24..31}
            acc[t] = __builtin_amdgcn_wmma_f32_16x16x32_bf16(
                false, af.v, false, bfr.v, (short)0, acc[t], false, false);
        }
        __syncthreads();
    }

    if constexpr (SOFTMAX_D) {
        // 64-row tile == one head (d=64). Softmax over rows, per column.
        __shared__ float sm[64 * 66];
        __shared__ float cmax[64];
        __shared__ float crinv[64];
#pragma unroll
        for (int t = 0; t < 4; ++t)
#pragma unroll
            for (int e = 0; e < 8; ++e) {
                const int row = wave * 16 + (lhalf ? e + 8 : e);
                const int col = t * 16 + lrow;
                sm[row * 66 + col] = acc[t][e] * scale;
            }
        __syncthreads();
        if (tid < 64) {
            float m = -3.0e38f;
            for (int r = 0; r < 64; ++r) m = fmaxf(m, sm[r * 66 + tid]);
            float s = 0.0f;
            for (int r = 0; r < 64; ++r) s += __expf(sm[r * 66 + tid] - m);
            cmax[tid] = m;
            crinv[tid] = 1.0f / s;
        }
        __syncthreads();
        for (int idx = tid; idx < 64 * 64; idx += 128) {
            const int r = idx >> 6, c = idx & 63;
            const float v = __expf(sm[r * 66 + c] - cmax[c]) * crinv[c];
            Ybf[(i64)b * sYb + (i64)(m0 + r) * ldy + n0 + c] = (bf16_t)v;
        }
    } else {
        float bv[8];
#pragma unroll
        for (int e = 0; e < 8; ++e) {
            const int row = m0 + wave * 16 + (lhalf ? e + 8 : e);
            bv[e] = bias ? bias[row] : 0.0f;
        }
#pragma unroll
        for (int t = 0; t < 4; ++t) {
            const int col = n0 + t * 16 + lrow;
#pragma unroll
            for (int e = 0; e < 8; ++e) {
                const int row = m0 + wave * 16 + (lhalf ? e + 8 : e);
                const float v = acc[t][e] + bv[e];
                const i64 off = (i64)b * sYb + (i64)row * ldy + col;
                if constexpr (WRITE_F32)  Yf[off] = v;
                if constexpr (WRITE_BF16)
                    if (row >= bfMinRow) Ybf[off] = (bf16_t)v;
            }
        }
    }
}

// ---------------------------------------------------------------------------
// Row softmax over N=4096 for K (rows 0..255 of the (B,512,4096) KV buffer).
// Reads f32 KV, writes normalized bf16 into the bf16 KV buffer (same rows).
// One 256-thread block per (c, b) row; 16 elements/thread.
// ---------------------------------------------------------------------------
__global__ __launch_bounds__(256)
void softmax_rows_k(const float* __restrict__ src, bf16_t* __restrict__ dst)
{
    const i64 base = (i64)blockIdx.y * (i64)(2 * Cc) * Nn + (i64)blockIdx.x * Nn;
    const float* s = src + base;
    __shared__ float red[256];
    const int tid = threadIdx.x;

    float vals[16];
    float m = -3.0e38f;
#pragma unroll
    for (int i = 0; i < 16; ++i) {
        vals[i] = s[tid + i * 256];
        m = fmaxf(m, vals[i]);
    }
    red[tid] = m;
    __syncthreads();
    for (int w = 128; w > 0; w >>= 1) {
        if (tid < w) red[tid] = fmaxf(red[tid], red[tid + w]);
        __syncthreads();
    }
    m = red[0];
    __syncthreads();

    float sum = 0.0f;
#pragma unroll
    for (int i = 0; i < 16; ++i) {
        vals[i] = __expf(vals[i] - m);
        sum += vals[i];
    }
    red[tid] = sum;
    __syncthreads();
    for (int w = 128; w > 0; w >>= 1) {
        if (tid < w) red[tid] += red[tid + w];
        __syncthreads();
    }
    const float rinv = 1.0f / red[0];
#pragma unroll
    for (int i = 0; i < 16; ++i)
        dst[base + tid + i * 256] = (bf16_t)(vals[i] * rinv);
}

// ---------------------------------------------------------------------------
extern "C" void kernel_launch(void* const* d_in, const int* in_sizes, int n_in,
                              void* d_out, int out_size, void* d_ws, size_t ws_size,
                              hipStream_t stream)
{
    (void)in_sizes; (void)n_in; (void)out_size; (void)ws_size;

    const float* x     = (const float*)d_in[0];   // (B, C, N) f32
    const float* cproj = (const float*)d_in[1];   // (B, C, N) f32
    const float* wq    = (const float*)d_in[2];   // (C, C)
    const float* bq    = (const float*)d_in[3];   // (C,) zeros; softmax-d path
    const float* wkv   = (const float*)d_in[4];   // (2C, C)
    const float* bkv   = (const float*)d_in[5];   // (2C,)
    const float* wo    = (const float*)d_in[6];   // (C, C)
    const float* bo    = (const float*)d_in[7];   // (C,)
    float*       out   = (float*)d_out;           // (B, C, N) f32
    (void)bq;

    // Workspace partition (~116 MB total)
    float*  kv_f = (float*)d_ws;                              // B*2C*N f32
    bf16_t* kv_b = (bf16_t*)(kv_f + (i64)Bb * 2 * Cc * Nn);   // B*2C*N bf16
    bf16_t* q_b  = kv_b + (i64)Bb * 2 * Cc * Nn;              // B*C*N bf16
    float*  ctx  = (float*)(q_b + (i64)Bb * Cc * Nn);         // B*C*C f32
    float*  mmat = ctx + (i64)Bb * Cc * Cc;                   // B*C*C f32

    const float scale = 0.25f; // 1/sqrt(sqrt(256))

    // 1) q = softmax_d(scale * (wq @ x))  -> q_b (bf16)
    gemm_wmma<float, float, /*TRANSB=*/false, /*SOFTMAX_D=*/true,
              /*F32=*/false, /*BF16=*/true>
        <<<dim3(Nn / 64, Cc / 64, Bb), 128, 0, stream>>>(
            wq, 0, Cc, x, (i64)Cc * Nn, Nn, nullptr,
            nullptr, q_b, (i64)Cc * Nn, Nn, Cc, scale, 0);

    // 2) KV = wkv @ cproj + bkv -> kv_f (f32, for k-softmax); bf16 copy only
    //    for the V half (rows >= C) -- K half is rewritten by step 3 anyway.
    gemm_wmma<float, float, false, false, /*F32=*/true, /*BF16=*/true>
        <<<dim3(Nn / 64, (2 * Cc) / 64, Bb), 128, 0, stream>>>(
            wkv, 0, Cc, cproj, (i64)Cc * Nn, Nn, bkv,
            kv_f, kv_b, (i64)2 * Cc * Nn, Nn, Cc, 1.0f, Cc);

    // 3) k = softmax_N(K) -> kv_b rows 0..C-1 (bf16)
    softmax_rows_k<<<dim3(Cc, Bb), 256, 0, stream>>>(kv_f, kv_b);

    // 4) context = k @ V^T  (M=C, N=C, K=4096; TRANSB: V rows contiguous in K)
    gemm_wmma<bf16_t, bf16_t, /*TRANSB=*/true, false, true, false>
        <<<dim3(Cc / 64, Cc / 64, Bb), 128, 0, stream>>>(
            kv_b, (i64)2 * Cc * Nn, Nn,                   // A = k
            kv_b + (i64)Cc * Nn, (i64)2 * Cc * Nn, Nn,    // B = V (NT)
            nullptr, ctx, nullptr, (i64)Cc * Cc, Cc, Nn, 1.0f, 0);

    // 5) Mmat = wo @ context  (folds the output projection: tiny GEMM)
    gemm_wmma<float, float, false, false, true, false>
        <<<dim3(Cc / 64, Cc / 64, Bb), 128, 0, stream>>>(
            wo, 0, Cc, ctx, (i64)Cc * Cc, Cc, nullptr,
            mmat, nullptr, (i64)Cc * Cc, Cc, Cc, 1.0f, 0);

    // 6) out = Mmat @ q + bo  -> d_out (f32)
    gemm_wmma<float, bf16_t, false, false, true, false>
        <<<dim3(Nn / 64, Cc / 64, Bb), 128, 0, stream>>>(
            mmat, (i64)Cc * Cc, Cc, q_b, (i64)Cc * Nn, Nn, bo,
            out, nullptr, (i64)Cc * Nn, Nn, Cc, 1.0f, 0);
}
